// BondingNetwork_45122926411833
// MI455X (gfx1250) — matrix operand order repaired and checked by reading, not compiled, not run on hardware
//
#include <hip/hip_runtime.h>
#include <hip/hip_bf16.h>

// ---------------- types for WMMA ----------------
typedef __attribute__((ext_vector_type(16))) __bf16 v16bf;  // A/B fragment, 16x16x32 bf16
typedef __attribute__((ext_vector_type(8)))  float  v8f;    // C/D fragment (f32)
typedef __attribute__((ext_vector_type(2)))  __bf16 v2bf;

// problem constants
#define BATCH 2
#define L 512
#define D 128                    // d_pair
#define NROWS (BATCH * L * L)    // 524288 MLP rows
#define PITCH 136                // padded LDS pitch in bf16 elems (272B -> bank-conflict-free)
#define TILES 4                  // 128-row tiles per block

__device__ __forceinline__ unsigned short f2bf(float f) {   // exact RNE (weights only)
    union { float f; unsigned int u; } x; x.f = f;
    unsigned int r = x.u + 0x7FFFu + ((x.u >> 16) & 1u);
    return (unsigned short)(r >> 16);
}

// pack two f32 -> packed bf16 (low = a, high = b).
// Fast path: single v_cvt_pk_bf16_f32 if the builtin exists.
// Fallback: round-to-nearest (ties away) adds + ONE v_perm_b32 packing both high halves.
__device__ __forceinline__ unsigned int pack2bf(float a, float b) {
#if __has_builtin(__builtin_amdgcn_cvt_pk_bf16_f32)
    union { v2bf v; unsigned int u; } c;
    c.v = __builtin_amdgcn_cvt_pk_bf16_f32(a, b);
    return c.u;
#else
    union { float f; unsigned int u; } xa, xb;
    xa.f = a; xb.f = b;
    const unsigned int au = xa.u + 0x8000u;
    const unsigned int bu = xb.u + 0x8000u;
    // result bytes (LSB..MSB): au.b2, au.b3, bu.b2, bu.b3  ->  {bf16(b), bf16(a)}
    return __builtin_amdgcn_perm(bu, au, 0x07060302u);
#endif
}

// ------------------------------------------------------------------
// Kernel 0: transpose + convert W1,W2 (f32 [k][n]) -> bf16 W^T [n][k]
// ------------------------------------------------------------------
__global__ void prep_weights_kernel(const float* __restrict__ W1,
                                    const float* __restrict__ W2,
                                    unsigned short* __restrict__ w1t,
                                    unsigned short* __restrict__ w2t) {
    for (int o = threadIdx.x; o < D * D; o += blockDim.x) {
        int n = o >> 7, k = o & 127;
        w1t[o] = f2bf(W1[k * D + n]);
        w2t[o] = f2bf(W2[k * D + n]);
    }
}

// ------------------------------------------------------------------
// Kernel 1: fused 3-layer MLP -> logits, bf16 WMMA, f32 accumulate.
// 256 threads = 8 waves, each wave owns a 16-row tile; TILES tiles/block.
// Weights live in padded LDS, filled via async global->LDS DMA.
// ------------------------------------------------------------------
__global__ __launch_bounds__(256) void mlp_logits_kernel(
    const float* __restrict__ pair,
    const unsigned short* __restrict__ w1t,
    const unsigned short* __restrict__ w2t,
    const float* __restrict__ b1, const float* __restrict__ b2,
    const float* __restrict__ W3, const float* __restrict__ b3,
    float* __restrict__ logits)
{
    extern __shared__ __align__(16) unsigned short smem[];
    unsigned short* w1s = smem;                    // 128 x PITCH
    unsigned short* w2s = smem + D * PITCH;        // 128 x PITCH
    unsigned short* astAll = smem + 2 * D * PITCH; // 8 waves x 16 x PITCH

    const int tid  = threadIdx.x;
    const int wave = tid >> 5;
    const int lane = tid & 31;
    const int r16  = lane & 15;          // row within wave tile (A/B lane pattern)
    const int hi   = lane >> 4;          // half-wave selector
    unsigned short* ast = astAll + wave * 16 * PITCH;

    // ---- async DMA: global bf16 W^T -> padded LDS (16B chunks per lane)
    {
        #pragma unroll
        for (int it = 0; it < 8; ++it) {
            const int g   = tid + it * 256;
            const int row = g >> 4, ci = g & 15;
            const unsigned long long src1 =
                (unsigned long long)(const void*)(w1t + row * D + ci * 8);
            const unsigned long long src2 =
                (unsigned long long)(const void*)(w2t + row * D + ci * 8);
            const unsigned dst1 = (unsigned)(unsigned long long)(void*)(w1s + row * PITCH + ci * 8);
            const unsigned dst2 = (unsigned)(unsigned long long)(void*)(w2s + row * PITCH + ci * 8);
            asm volatile("global_load_async_to_lds_b128 %0, %1, off"
                         :: "v"(dst1), "v"(src1) : "memory");
            asm volatile("global_load_async_to_lds_b128 %0, %1, off"
                         :: "v"(dst2), "v"(src2) : "memory");
        }
        asm volatile("s_wait_asynccnt 0" ::: "memory");
        __syncthreads();
    }

    const int kstart = hi * 8;   // B lane pattern: K = 8*hi+[0,8) and 16+8*hi+[0,8) per lane
    union AF { v16bf v; unsigned short s[16]; unsigned int u[8]; uint4 q[2]; };

    for (int t = 0; t < TILES; ++t) {
        const int rowBase = ((blockIdx.x * TILES + t) * 8 + wave) * 16;

        // ---- A fragments for layer 1 straight from global pair rows (f32 -> bf16)
        AF a1[4];
        const float* arow = pair + (size_t)(rowBase + r16) * D;
        #pragma unroll
        for (int kb = 0; kb < 4; ++kb) {
            const int k0 = kb * 32 + hi * 8;
            float4 f0 = *(const float4*)(arow + k0);
            float4 f1 = *(const float4*)(arow + k0 + 4);
            float4 f2 = *(const float4*)(arow + k0 + 16);
            float4 f3 = *(const float4*)(arow + k0 + 20);
            a1[kb].u[0] = pack2bf(f0.x, f0.y);
            a1[kb].u[1] = pack2bf(f0.z, f0.w);
            a1[kb].u[2] = pack2bf(f1.x, f1.y);
            a1[kb].u[3] = pack2bf(f1.z, f1.w);
            a1[kb].u[4] = pack2bf(f2.x, f2.y);
            a1[kb].u[5] = pack2bf(f2.z, f2.w);
            a1[kb].u[6] = pack2bf(f3.x, f3.y);
            a1[kb].u[7] = pack2bf(f3.z, f3.w);
        }

        // ---- layer 1: 8 column tiles x 4 K-steps, B from LDS (conflict-free)
        #pragma unroll
        for (int nt = 0; nt < 8; ++nt) {
            const int col = nt * 16 + r16;
            AF bfr[4];                       // preload all 4 B fragments (staggered waits)
            #pragma unroll
            for (int kb = 0; kb < 4; ++kb) {
                const unsigned short* bp = w1s + col * PITCH + kb * 32 + kstart;
                bfr[kb].q[0] = *(const uint4*)(const void*)(bp);
                bfr[kb].q[1] = *(const uint4*)(const void*)(bp + 16);
            }
            const float bias = b1[col];
            v8f c;
            #pragma unroll
            for (int i = 0; i < 8; ++i) c[i] = bias;
            #pragma unroll
            for (int kb = 0; kb < 4; ++kb)
                c = __builtin_amdgcn_wmma_f32_16x16x32_bf16(false, a1[kb].v, false, bfr[kb].v,
                                                            (short)0, c, false, false);
            // relu -> packed bf16 -> LDS stage (C layout: VGPR i = row i+8*hi, lane&15 = col)
            #pragma unroll
            for (int i = 0; i < 8; i += 2) {
                const unsigned int pk = pack2bf(fmaxf(c[i], 0.f), fmaxf(c[i + 1], 0.f));
                ast[(i + hi * 8) * PITCH + col]     = (unsigned short)pk;
                ast[(i + 1 + hi * 8) * PITCH + col] = (unsigned short)(pk >> 16);
            }
        }
        asm volatile("s_wait_dscnt 0" ::: "memory");  // same-wave LDS store->load fence

        // ---- A fragments for layer 2 from padded LDS staging
        AF a2[4];
        #pragma unroll
        for (int kb = 0; kb < 4; ++kb) {
            const int k0 = kb * 32 + hi * 8;
            a2[kb].q[0] = *(const uint4*)(const void*)(ast + r16 * PITCH + k0);
            a2[kb].q[1] = *(const uint4*)(const void*)(ast + r16 * PITCH + k0 + 16);
        }

        // ---- layer 2 + fused layer-3 dot with W3
        float p[8];
        #pragma unroll
        for (int i = 0; i < 8; ++i) p[i] = 0.f;

        #pragma unroll
        for (int nt = 0; nt < 8; ++nt) {
            const int col = nt * 16 + r16;
            AF bfr[4];
            #pragma unroll
            for (int kb = 0; kb < 4; ++kb) {
                const unsigned short* bp = w2s + col * PITCH + kb * 32 + kstart;
                bfr[kb].q[0] = *(const uint4*)(const void*)(bp);
                bfr[kb].q[1] = *(const uint4*)(const void*)(bp + 16);
            }
            const float bias = b2[col];
            v8f c;
            #pragma unroll
            for (int i = 0; i < 8; ++i) c[i] = bias;
            #pragma unroll
            for (int kb = 0; kb < 4; ++kb)
                c = __builtin_amdgcn_wmma_f32_16x16x32_bf16(false, a2[kb].v, false, bfr[kb].v,
                                                            (short)0, c, false, false);
            const float w3c = W3[col];
            #pragma unroll
            for (int i = 0; i < 8; ++i)
                p[i] = fmaf(fmaxf(c[i], 0.f), w3c, p[i]);
        }

        // butterfly reduce over the 16 lanes of each half (rows 0-7 / 8-15)
        #pragma unroll
        for (int m = 1; m < 16; m <<= 1) {
            #pragma unroll
            for (int i = 0; i < 8; ++i) p[i] += __shfl_xor(p[i], m, 32);
        }

        if (r16 == 0) {
            const float bb = b3[0];
            #pragma unroll
            for (int i = 0; i < 8; ++i)
                logits[rowBase + hi * 8 + i] = p[i] + bb;
        }
        asm volatile("s_wait_dscnt 0" ::: "memory");  // WAR fence before next tile's staging
    }
}

// ------------------------------------------------------------------
// Kernel 2: per-row max of logits. Grid = B*L blocks of 256.
// ------------------------------------------------------------------
__global__ void rowmax_kernel(const float* __restrict__ logits, float* __restrict__ rmax) {
    __shared__ float sm[256];
    const int r = blockIdx.x;
    const float* row = logits + (size_t)r * L;
    const int t = threadIdx.x;
    float m = fmaxf(row[t], row[t + 256]);
    sm[t] = m; __syncthreads();
    for (int s = 128; s > 0; s >>= 1) {
        if (t < s) sm[t] = fmaxf(sm[t], sm[t + s]);
        __syncthreads();
    }
    if (t == 0) rmax[r] = sm[0];
}

// ------------------------------------------------------------------
// Kernel 3: M0 = mask ? exp(2*((lij-ri)+(lji-rj))) : 0 ; also emits
// initial column sums (axis=-2). Block owns 16 columns. Grid = B*32.
// ------------------------------------------------------------------
__global__ void build_m0_kernel(const float* __restrict__ logits,
                                const float* __restrict__ rmax,
                                const unsigned char* __restrict__ mask,
                                float* __restrict__ mbuf, float* __restrict__ colsum) {
    __shared__ float sm[256];
    const int b  = blockIdx.x >> 5;
    const int j0 = (blockIdx.x & 31) * 16;
    const int tx = threadIdx.x & 15;
    const int ty = threadIdx.x >> 4;
    const int j  = j0 + tx;
    const size_t base = (size_t)b * L * L;
    const float rmj = rmax[b * L + j];
    float acc = 0.f;
    for (int i = ty; i < L; i += 16) {
        size_t idx = base + (size_t)i * L + j;
        float cij = logits[idx] - rmax[b * L + i];
        float cji = logits[base + (size_t)j * L + i] - rmj;
        float v = mask[idx] ? __expf((cij + cji) * 2.0f) : 0.0f;  // (sym/2)/tau, tau=0.25
        mbuf[idx] = v;
        acc += v;
    }
    sm[threadIdx.x] = acc; __syncthreads();
    for (int s = 128; s >= 16; s >>= 1) {
        if (threadIdx.x < s) sm[threadIdx.x] += sm[threadIdx.x + s];
        __syncthreads();
    }
    if (threadIdx.x < 16) colsum[b * L + j0 + threadIdx.x] = sm[threadIdx.x];
}

// ------------------------------------------------------------------
// Kernel 4: m /= clip(colsum_j); m *= mask; emit rowsum_i (axis=-1).
// Block owns one row. Grid = B*L.
// ------------------------------------------------------------------
__global__ void colnorm_rowsum_kernel(float* __restrict__ mbuf,
                                      const float* __restrict__ colsum,
                                      const unsigned char* __restrict__ mask,
                                      float* __restrict__ rowsum) {
    __shared__ float sm[256];
    const int r = blockIdx.x;             // b*L + i
    const int b = r >> 9;
    float* row = mbuf + (size_t)r * L;
    const unsigned char* mrow = mask + (size_t)r * L;
    const float* cs = colsum + b * L;
    const int t = threadIdx.x;
    float acc = 0.f;
    #pragma unroll
    for (int s = 0; s < 2; ++s) {
        int j = t + s * 256;
        float v = row[j] / fmaxf(cs[j], 1e-8f);
        v *= (float)mrow[j];
        row[j] = v;
        acc += v;
    }
    sm[t] = acc; __syncthreads();
    for (int s = 128; s > 0; s >>= 1) {
        if (t < s) sm[t] += sm[t + s];
        __syncthreads();
    }
    if (t == 0) rowsum[r] = sm[0];
}

// ------------------------------------------------------------------
// Kernel 5: m /= clip(rowsum_i); m *= mask; emit colsum_j (axis=-2).
// Block owns 16 columns. Grid = B*32.
// ------------------------------------------------------------------
__global__ void rownorm_colsum_kernel(float* __restrict__ mbuf,
                                      const float* __restrict__ rowsum,
                                      const unsigned char* __restrict__ mask,
                                      float* __restrict__ colsum) {
    __shared__ float sm[256];
    const int b  = blockIdx.x >> 5;
    const int j0 = (blockIdx.x & 31) * 16;
    const int tx = threadIdx.x & 15;
    const int ty = threadIdx.x >> 4;
    const int j  = j0 + tx;
    const size_t base = (size_t)b * L * L;
    float acc = 0.f;
    for (int i = ty; i < L; i += 16) {
        size_t idx = base + (size_t)i * L + j;
        float v = mbuf[idx] / fmaxf(rowsum[b * L + i], 1e-8f);
        v *= (float)mask[idx];
        mbuf[idx] = v;
        acc += v;
    }
    sm[threadIdx.x] = acc; __syncthreads();
    for (int s = 128; s >= 16; s >>= 1) {
        if (threadIdx.x < s) sm[threadIdx.x] += sm[threadIdx.x + s];
        __syncthreads();
    }
    if (threadIdx.x < 16) colsum[b * L + j0 + threadIdx.x] = sm[threadIdx.x];
}

// ------------------------------------------------------------------
// Kernel 6: out = 0.5*(m + m^T) * mask. Grid = B*L.
// ------------------------------------------------------------------
__global__ void final_sym_kernel(const float* __restrict__ mbuf,
                                 const unsigned char* __restrict__ mask,
                                 float* __restrict__ out) {
    const int r = blockIdx.x;
    const int b = r >> 9, i = r & 511;
    const size_t base = (size_t)b * L * L;
    const int t = threadIdx.x;
    #pragma unroll
    for (int s = 0; s < 2; ++s) {
        int j = t + s * 256;
        size_t idx = base + (size_t)i * L + j;
        float v = (mbuf[idx] + mbuf[base + (size_t)j * L + i]) * 0.5f;
        out[idx] = v * (float)mask[idx];
    }
}

// ------------------------------------------------------------------
extern "C" void kernel_launch(void* const* d_in, const int* in_sizes, int n_in,
                              void* d_out, int out_size, void* d_ws, size_t ws_size,
                              hipStream_t stream) {
    (void)in_sizes; (void)n_in; (void)out_size; (void)ws_size;
    const float* pair        = (const float*)d_in[2];
    const unsigned char* msk = (const unsigned char*)d_in[3];   // bool array (1 byte/elem)
    const float* W1 = (const float*)d_in[4];
    const float* b1 = (const float*)d_in[5];
    const float* W2 = (const float*)d_in[6];
    const float* b2 = (const float*)d_in[7];
    const float* W3 = (const float*)d_in[8];
    const float* b3 = (const float*)d_in[9];
    float* out = (float*)d_out;

    char* ws = (char*)d_ws;
    float* logits = (float*)ws;  ws += (size_t)NROWS * 4;
    float* mbuf   = (float*)ws;  ws += (size_t)NROWS * 4;
    float* rmax   = (float*)ws;  ws += (size_t)BATCH * L * 4;
    float* colsum = (float*)ws;  ws += (size_t)BATCH * L * 4;
    float* rowsum = (float*)ws;  ws += (size_t)BATCH * L * 4;
    unsigned short* w1t = (unsigned short*)ws;  ws += (size_t)D * D * 2;
    unsigned short* w2t = (unsigned short*)ws;  ws += (size_t)D * D * 2;

    const size_t lds_bytes = (size_t)(2 * D * PITCH + 8 * 16 * PITCH) * 2;  // 104448 B

    prep_weights_kernel<<<1, 256, 0, stream>>>(W1, W2, w1t, w2t);
    mlp_logits_kernel<<<NROWS / (128 * TILES), 256, lds_bytes, stream>>>(
        pair, w1t, w2t, b1, b2, W3, b3, logits);
    rowmax_kernel<<<BATCH * L, 256, 0, stream>>>(logits, rmax);
    build_m0_kernel<<<BATCH * 32, 256, 0, stream>>>(logits, rmax, msk, mbuf, colsum);
    for (int it = 0; it < 30; ++it) {
        colnorm_rowsum_kernel<<<BATCH * L, 256, 0, stream>>>(mbuf, colsum, msk, rowsum);
        rownorm_colsum_kernel<<<BATCH * 32, 256, 0, stream>>>(mbuf, rowsum, msk, colsum);
    }
    final_sym_kernel<<<BATCH * L, 256, 0, stream>>>(mbuf, msk, out);
}
